// RingAttractorNetwork_38139309588533
// MI455X (gfx1250) — matrix operand order repaired and checked by reading, not compiled, not run on hardware
//
#include <hip/hip_runtime.h>

typedef __attribute__((ext_vector_type(16))) _Float16 v16h;
typedef __attribute__((ext_vector_type(8)))  _Float16 v8h;
typedef __attribute__((ext_vector_type(8)))  float    v8f;

#define N_NEUR  256
#define T_STEPS 512
#define B_BATCH 1024
#define M_ROWS  32      // batch rows per workgroup
#define LDK     264     // padded LDS row length in f16 elements (132 DW stride == 4 mod 64 banks)
#define DT_C    0.01f

static __device__ __forceinline__ v16h cat8(v8h lo, v8h hi) {
  return __builtin_shufflevector(lo, hi, 0,1,2,3,4,5,6,7,8,9,10,11,12,13,14,15);
}

__global__ __launch_bounds__(256) void ring_attractor_wmma(
    const float* __restrict__ W,          // [256,256] row = presyn k, col = postsyn n
    const float* __restrict__ log_scale,  // [1]
    const float* __restrict__ gain,       // [256]
    const float* __restrict__ bias,       // [256]
    const float* __restrict__ log_tau,    // [256]
    const float* __restrict__ vscale_p,   // [1]
    const float* __restrict__ lscale_p,   // [1]
    const float* __restrict__ velocity,   // [512,1024,2]
    const float* __restrict__ landmarks,  // [512,1024,64]
    float* __restrict__ out)              // [1024,64]
{
  // W transposed to [n][k] so B fragments are contiguous-per-lane; state [m][k].
  __shared__ __attribute__((aligned(32))) _Float16 sWt[N_NEUR * LDK];
  __shared__ __attribute__((aligned(32))) _Float16 sS [M_ROWS * LDK];

  const int tid  = threadIdx.x;
  const int wg   = blockIdx.x;
  const int wave = tid >> 5;
  const int lane = tid & 31;
  const int l    = lane & 15;
  const int half = lane >> 4;

  // ---------------- one-time setup ----------------
  const float esc = __expf(log_scale[0]);
  for (int idx = tid; idx < N_NEUR * N_NEUR; idx += 256) {
    const int k = idx >> 8;       // row of W (presyn)
    const int n = idx & 255;      // col of W (postsyn)
    sWt[n * LDK + k] = (_Float16)(esc * W[idx]);
  }
  for (int idx = tid; idx < M_ROWS * LDK; idx += 256) sS[idx] = (_Float16)0.f;

  // wave -> output tiles: N-tiles {2w, 2w+1}, both M-tiles {0,1}
  const int nt0 = wave * 2;
  const int nt1 = wave * 2 + 1;
  const int ng[2]   = { nt0 * 16 + l, nt1 * 16 + l };
  const float vsc = vscale_p[0];
  const float lsc = lscale_p[0];
  float g_[2], b_[2], cf[2];
  int mode[2];                    // 0=landmark, 1=vel.x, 2=vel.y, 3=none (tile-uniform)
  #pragma unroll
  for (int j = 0; j < 2; ++j) {
    g_[j] = gain[ng[j]];
    b_[j] = bias[ng[j]];
    cf[j] = DT_C * __expf(-log_tau[ng[j]]);
    mode[j] = (ng[j] < 64) ? 0 : (ng[j] < 96) ? 1 : (ng[j] < 128) ? 2 : 3;
  }

  const int rowbase = half * 8;               // local batch-row base for C layout
  v8f s[4] = { v8f{0.f}, v8f{0.f}, v8f{0.f}, v8f{0.f} };  // exact f32 state, tile (m*2+j)

  __syncthreads();

  // ---------------- 512 Euler steps ----------------
  #pragma unroll 1
  for (int t = 0; t < T_STEPS; ++t) {
    // prefetch external input for this step (independent of state)
    float ext[2][2][8];
    const float* lm  = landmarks + (size_t)t * (B_BATCH * 64);
    const float* vel = velocity  + (size_t)t * (B_BATCH * 2);
    #pragma unroll
    for (int m = 0; m < 2; ++m) {
      #pragma unroll
      for (int j = 0; j < 2; ++j) {
        if (mode[j] == 0) {
          #pragma unroll
          for (int c = 0; c < 8; ++c) {
            const int b = wg * M_ROWS + m * 16 + rowbase + c;
            ext[m][j][c] = lm[(size_t)b * 64 + ng[j]] * lsc;
          }
        } else if (mode[j] == 1) {
          #pragma unroll
          for (int c = 0; c < 8; ++c) {
            const int b = wg * M_ROWS + m * 16 + rowbase + c;
            ext[m][j][c] = vel[(size_t)b * 2 + 0] * vsc;
          }
        } else if (mode[j] == 2) {
          #pragma unroll
          for (int c = 0; c < 8; ++c) {
            const int b = wg * M_ROWS + m * 16 + rowbase + c;
            ext[m][j][c] = vel[(size_t)b * 2 + 1] * vsc;
          }
        } else {
          #pragma unroll
          for (int c = 0; c < 8; ++c) ext[m][j][c] = 0.f;
        }
      }
    }

    // GEMM: total = S @ W_eff  (f16 in, f32 accumulate)
    v8f acc[4] = { v8f{0.f}, v8f{0.f}, v8f{0.f}, v8f{0.f} };
    #pragma unroll
    for (int kt = 0; kt < 8; ++kt) {
      // A fragments (16x32 f16): lane<16 -> K{0..7,16..23}; lane>=16 -> K{8..15,24..31}
      const int ka = kt * 32 + half * 8;
      const v8h a0lo = *(const v8h*)&sS[(0 * 16 + l) * LDK + ka];
      const v8h a0hi = *(const v8h*)&sS[(0 * 16 + l) * LDK + ka + 16];
      const v8h a1lo = *(const v8h*)&sS[(1 * 16 + l) * LDK + ka];
      const v8h a1hi = *(const v8h*)&sS[(1 * 16 + l) * LDK + ka + 16];
      const v16h A0 = cat8(a0lo, a0hi);
      const v16h A1 = cat8(a1lo, a1hi);
      // B fragments (32x16 f16): lane<16 -> K{0..7,16..23} at column N=l; lane>=16 -> K{8..15,24..31}
      const int kb = kt * 32 + half * 8;
      const v8h b0lo = *(const v8h*)&sWt[(nt0 * 16 + l) * LDK + kb];
      const v8h b0hi = *(const v8h*)&sWt[(nt0 * 16 + l) * LDK + kb + 16];
      const v8h b1lo = *(const v8h*)&sWt[(nt1 * 16 + l) * LDK + kb];
      const v8h b1hi = *(const v8h*)&sWt[(nt1 * 16 + l) * LDK + kb + 16];
      const v16h B0 = cat8(b0lo, b0hi);
      const v16h B1 = cat8(b1lo, b1hi);

      acc[0] = __builtin_amdgcn_wmma_f32_16x16x32_f16(false, A0, false, B0, (short)0, acc[0], false, false);
      acc[1] = __builtin_amdgcn_wmma_f32_16x16x32_f16(false, A0, false, B1, (short)0, acc[1], false, false);
      acc[2] = __builtin_amdgcn_wmma_f32_16x16x32_f16(false, A1, false, B0, (short)0, acc[2], false, false);
      acc[3] = __builtin_amdgcn_wmma_f32_16x16x32_f16(false, A1, false, B1, (short)0, acc[3], false, false);
    }

    __syncthreads();   // all reads of sS complete before overwrite

    // elementwise: rate = relu(gain*(total+ext)+bias); s += coef*(rate - s)
    #pragma unroll
    for (int m = 0; m < 2; ++m) {
      #pragma unroll
      for (int j = 0; j < 2; ++j) {
        const int ti = m * 2 + j;
        #pragma unroll
        for (int c = 0; c < 8; ++c) {
          const float tv   = acc[ti][c] + ext[m][j][c];
          const float rate = fmaxf(0.f, g_[j] * tv + b_[j]);
          float sv = s[ti][c];
          sv += cf[j] * (rate - sv);
          s[ti][c] = sv;
          sS[(m * 16 + rowbase + c) * LDK + ng[j]] = (_Float16)sv;
        }
      }
    }

    __syncthreads();   // new state visible before next step's reads
  }

  // ---------------- output: epg slice, columns 0..63 (waves 0 and 1) ----------------
  if (wave < 2) {
    #pragma unroll
    for (int m = 0; m < 2; ++m) {
      #pragma unroll
      for (int j = 0; j < 2; ++j) {
        #pragma unroll
        for (int c = 0; c < 8; ++c) {
          const int b = wg * M_ROWS + m * 16 + rowbase + c;
          out[(size_t)b * 64 + ng[j]] = s[m * 2 + j][c];
        }
      }
    }
  }
}

extern "C" void kernel_launch(void* const* d_in, const int* in_sizes, int n_in,
                              void* d_out, int out_size, void* d_ws, size_t ws_size,
                              hipStream_t stream) {
  (void)in_sizes; (void)n_in; (void)d_ws; (void)ws_size; (void)out_size;
  const float* W         = (const float*)d_in[0];
  const float* log_scale = (const float*)d_in[1];
  const float* gain      = (const float*)d_in[2];
  const float* bias      = (const float*)d_in[3];
  const float* log_tau   = (const float*)d_in[4];
  const float* vscale    = (const float*)d_in[5];
  const float* lscale    = (const float*)d_in[6];
  const float* velocity  = (const float*)d_in[7];
  const float* landmarks = (const float*)d_in[8];
  float* out = (float*)d_out;

  dim3 grid(B_BATCH / M_ROWS);   // 32 workgroups
  dim3 block(256);               // 8 wave32 waves
  ring_attractor_wmma<<<grid, block, 0, stream>>>(
      W, log_scale, gain, bias, log_tau, vscale, lscale, velocity, landmarks, out);
}
